// LSTMSeq2SeqPredictor_11098195493388
// MI455X (gfx1250) — matrix operand order repaired and checked by reading, not compiled, not run on hardware
//
#include <hip/hip_runtime.h>

typedef __bf16 bf16_t;
typedef __bf16 v8bf  __attribute__((ext_vector_type(8)));
typedef __bf16 v16bf __attribute__((ext_vector_type(16)));
typedef float  v8f   __attribute__((ext_vector_type(8)));

// Problem constants
static constexpr int Hh    = 64;    // hidden
static constexpr int NIN   = 9;     // input features
static constexpr int SRCL  = 7;
static constexpr int PREDL = 10;

// LDS byte offsets (total 151616 B; 2 blocks/WGP within 320 KB)
static constexpr int OFF_EWHH = 0;       // 16 tiles * 2 ks * 32 lanes * 16 halves * 2B = 32768
static constexpr int OFF_DWHH = 32768;   // 32768
static constexpr int OFF_EWIH = 65536;   // 16 * 32 * 16 * 2 = 16384
static constexpr int OFF_DWIH = 81920;   // 16384
static constexpr int OFF_FC   = 98304;   // 2 * 32 * 16 * 2 = 2048
static constexpr int OFF_X    = 100352;  // 128 rows * 7 t * 16 halves * 2 = 28672
static constexpr int OFF_HS   = 129024;  // 8 waves * 16 * 64 * 2 = 16384
static constexpr int OFF_PS   = 145408;  // 8 waves * 16 * 16 * 2 = 4096
static constexpr int OFF_BE   = 149504;  // 256 f32 = 1024
static constexpr int OFF_BD   = 150528;  // 1024
static constexpr int OFF_FB   = 151552;  // 16 f32 = 64
static constexpr int LDS_BYTES = 151616;

__device__ __forceinline__ float sigmoidf_(float x) {
    return 1.0f / (1.0f + __expf(-x));
}
__device__ __forceinline__ float tanhf_(float x) {
    return 1.0f - 2.0f / (__expf(2.0f * x) + 1.0f);
}

__global__ void __launch_bounds__(256)
lstm_s2s_wmma(const float* __restrict__ src,
              const float* __restrict__ eWih, const float* __restrict__ eWhh, const float* __restrict__ eB,
              const float* __restrict__ dWih, const float* __restrict__ dWhh, const float* __restrict__ dB,
              const float* __restrict__ fcW,  const float* __restrict__ fcB,
              float* __restrict__ out)
{
    extern __shared__ char sm[];
    const int tid  = threadIdx.x;
    const int lane = tid & 31;
    const int wave = tid >> 5;
    const int wg   = blockIdx.x;

    // ---------------- Prologue: swizzle weights into WMMA B-fragment order ----------------
    // B frag as packed here (bf16 KxN=32x16, contiguous K matching the A loads): lane L holds N=L&15, half i holds K=(L>>4)*16+i.
    // Whh (256x64): frag-lane rows: fl = t*64 + ks*32 + l  (1024 per matrix, 4/thread)
    for (int m = 0; m < 2; ++m) {
        const float* W = m ? dWhh : eWhh;
        bf16_t* dst = (bf16_t*)(sm + (m ? OFF_DWHH : OFF_EWHH));
        for (int f = 0; f < 4; ++f) {
            int fl = tid * 4 + f;
            int t  = fl >> 6;
            int ks = (fl >> 5) & 1;
            int l  = fl & 31;
            int N  = t * 16 + (l & 15);
            int Kb = ks * 32 + ((l >> 4) << 4);
            bf16_t* p = dst + fl * 16;
            for (int i = 0; i < 16; ++i) p[i] = (bf16_t)W[N * Hh + Kb + i];
        }
    }
    // Wih (256x9), K padded to 32: fl = t*32 + l (512 per matrix, 2/thread)
    for (int m = 0; m < 2; ++m) {
        const float* W = m ? dWih : eWih;
        bf16_t* dst = (bf16_t*)(sm + (m ? OFF_DWIH : OFF_EWIH));
        for (int f = 0; f < 2; ++f) {
            int fl = tid * 2 + f;
            int t  = fl >> 5;
            int l  = fl & 31;
            int N  = t * 16 + (l & 15);
            int Kb = (l >> 4) << 4;
            bf16_t* p = dst + fl * 16;
            for (int i = 0; i < 16; ++i) {
                int K = Kb + i;
                p[i] = (K < NIN) ? (bf16_t)W[N * NIN + K] : (bf16_t)0.0f;
            }
        }
    }
    // fc_W (9x64), N padded to 16: 64 frag-lanes
    if (tid < 64) {
        int ks = tid >> 5, l = tid & 31;
        int N  = l & 15;
        int Kb = ks * 32 + ((l >> 4) << 4);
        bf16_t* p = (bf16_t*)(sm + OFF_FC) + tid * 16;
        for (int i = 0; i < 16; ++i)
            p[i] = (N < NIN) ? (bf16_t)fcW[N * Hh + Kb + i] : (bf16_t)0.0f;
    }
    // biases
    ((float*)(sm + OFF_BE))[tid] = eB[tid];
    ((float*)(sm + OFF_BD))[tid] = dB[tid];
    if (tid < 16) ((float*)(sm + OFF_FB))[tid] = (tid < NIN) ? fcB[tid] : 0.0f;
    // stage src slice as bf16, K padded to 16: [r][t][16]
    {
        bf16_t* x = (bf16_t*)(sm + OFF_X);
        for (int e = tid; e < 128 * SRCL * 16; e += 256) {
            int k = e & 15;
            int t = (e >> 4) % SRCL;
            int r = e / (16 * SRCL);
            float v = (k < NIN) ? src[((size_t)(wg * 128 + r) * SRCL + t) * NIN + k] : 0.0f;
            x[e] = (bf16_t)v;
        }
    }
    // zero h staging (h0 = 0) and pred staging (pad columns must be 0)
    for (int e = tid; e < 8 * 16 * Hh; e += 256) ((bf16_t*)(sm + OFF_HS))[e] = (bf16_t)0.0f;
    for (int e = tid; e < 8 * 16 * 16; e += 256) ((bf16_t*)(sm + OFF_PS))[e] = (bf16_t)0.0f;
    __syncthreads();

    // ---------------- Main loop (all staging below is wave-private; no barriers) ----------------
    const int M16    = lane & 15;      // A: row M ; B/C/D: col N
    const int hiHalf = lane >> 4;
    const int ko     = hiHalf * 16;    // A-frag K sub-offset

    bf16_t* hbase = (bf16_t*)(sm + OFF_HS) + wave * (16 * Hh);
    bf16_t* pbase = (bf16_t*)(sm + OFF_PS) + wave * (16 * 16);
    const bf16_t* xl = (const bf16_t*)(sm + OFF_X);

    // A frag of h (16x32 bf16, K window kw): two contiguous 16B LDS reads per lane
    auto loadA_h = [&](int kw) -> v16bf {
        v8bf lo = *(const v8bf*)(hbase + M16 * Hh + kw * 32 + ko);
        v8bf hi = *(const v8bf*)(hbase + M16 * Hh + kw * 32 + ko + 8);
        return __builtin_shufflevector(lo, hi, 0,1,2,3,4,5,6,7,8,9,10,11,12,13,14,15);
    };
    // A frag of a 16-half padded input row (K=0..15 real, 16..31 zero)
    auto loadA_row16 = [&](const bf16_t* rowbase) -> v16bf {
        v8bf lo = *(const v8bf*)(rowbase);
        v8bf hi = *(const v8bf*)(rowbase + 8);
        if (hiHalf) for (int i = 0; i < 8; ++i) { lo[i] = (bf16_t)0.0f; hi[i] = (bf16_t)0.0f; }
        return __builtin_shufflevector(lo, hi, 0,1,2,3,4,5,6,7,8,9,10,11,12,13,14,15);
    };

    float c[4][8];
    for (int kt = 0; kt < 4; ++kt)
        for (int j = 0; j < 8; ++j) c[kt][j] = 0.0f;

    auto lstm_step = [&](int whhOff, int wihOff, int bOff, const bf16_t* xrow) {
        v16bf hA0 = loadA_h(0);
        v16bf hA1 = loadA_h(1);
        v16bf xA  = loadA_row16(xrow);
        for (int kt = 0; kt < 4; ++kt) {
            v8f acc[4];
            for (int q = 0; q < 4; ++q) {           // q: i,f,g,o gate chunk
                int t = q * 4 + kt;                 // gate tile index (N window)
                float b = ((const float*)(sm + bOff))[t * 16 + M16];
                v8f a;
                for (int j = 0; j < 8; ++j) a[j] = b;
                v16bf bw = *(const v16bf*)(sm + wihOff + (t * 32 + lane) * 32);
                a = __builtin_amdgcn_wmma_f32_16x16x32_bf16(false, xA, false, bw, (short)0, a, false, false);
                v16bf b0 = *(const v16bf*)(sm + whhOff + (((t * 2 + 0) * 32) + lane) * 32);
                a = __builtin_amdgcn_wmma_f32_16x16x32_bf16(false, hA0, false, b0, (short)0, a, false, false);
                v16bf b1 = *(const v16bf*)(sm + whhOff + (((t * 2 + 1) * 32) + lane) * 32);
                a = __builtin_amdgcn_wmma_f32_16x16x32_bf16(false, hA1, false, b1, (short)0, a, false, false);
                acc[q] = a;
            }
            for (int j = 0; j < 8; ++j) {
                float ig = sigmoidf_(acc[0][j]);
                float fg = sigmoidf_(acc[1][j]);
                float gg = tanhf_(acc[2][j]);
                float og = sigmoidf_(acc[3][j]);
                float cn = fg * c[kt][j] + ig * gg;
                c[kt][j] = cn;
                float hn = og * tanhf_(cn);
                int M = j + hiHalf * 8;
                hbase[M * Hh + kt * 16 + M16] = (bf16_t)hn;  // restage new h (bf16)
            }
        }
    };

    // Encoder
    for (int t = 0; t < SRCL; ++t) {
        const bf16_t* xrow = xl + ((size_t)(wave * 16 + M16) * SRCL + t) * 16;
        lstm_step(OFF_EWHH, OFF_EWIH, OFF_BE, xrow);
    }

    // Decoder (autoregressive)
    const size_t row0 = (size_t)wg * 128 + wave * 16;
    for (int d = 0; d < PREDL; ++d) {
        const bf16_t* xrow = (d == 0)
            ? xl + ((size_t)(wave * 16 + M16) * SRCL + (SRCL - 1)) * 16
            : pbase + M16 * 16;
        lstm_step(OFF_DWHH, OFF_DWIH, OFF_BD, xrow);

        // pred = h_new @ fc_W^T + fc_b   (N padded 9->16)
        v16bf hA0 = loadA_h(0);
        v16bf hA1 = loadA_h(1);
        v8f p;
        {
            float fb = ((const float*)(sm + OFF_FB))[M16];
            for (int j = 0; j < 8; ++j) p[j] = fb;
        }
        v16bf f0 = *(const v16bf*)(sm + OFF_FC + (0 * 32 + lane) * 32);
        p = __builtin_amdgcn_wmma_f32_16x16x32_bf16(false, hA0, false, f0, (short)0, p, false, false);
        v16bf f1 = *(const v16bf*)(sm + OFF_FC + (1 * 32 + lane) * 32);
        p = __builtin_amdgcn_wmma_f32_16x16x32_bf16(false, hA1, false, f1, (short)0, p, false, false);

        for (int j = 0; j < 8; ++j) {
            int M = j + hiHalf * 8;
            pbase[M * 16 + M16] = (bf16_t)p[j];            // restage as next input
            if (M16 < NIN)
                out[((row0 + M) * PREDL + d) * NIN + M16] = p[j];
        }
    }
}

extern "C" void kernel_launch(void* const* d_in, const int* in_sizes, int n_in,
                              void* d_out, int out_size, void* d_ws, size_t ws_size,
                              hipStream_t stream) {
    const float* src  = (const float*)d_in[0];
    const float* eWih = (const float*)d_in[1];
    const float* eWhh = (const float*)d_in[2];
    const float* eB   = (const float*)d_in[3];
    const float* dWih = (const float*)d_in[4];
    const float* dWhh = (const float*)d_in[5];
    const float* dB   = (const float*)d_in[6];
    const float* fcW  = (const float*)d_in[7];
    const float* fcB  = (const float*)d_in[8];
    float* out = (float*)d_out;

    const int batch = in_sizes[0] / (SRCL * NIN);   // 65536
    const int grid  = batch / 128;                  // 128 rows per block (8 waves x 16)

    (void)hipFuncSetAttribute((const void*)lstm_s2s_wmma,
                              hipFuncAttributeMaxDynamicSharedMemorySize, LDS_BYTES);
    lstm_s2s_wmma<<<grid, 256, LDS_BYTES, stream>>>(src, eWih, eWhh, eB,
                                                    dWih, dWhh, dB, fcW, fcB, out);
}
